// AoE_17738214933201
// MI455X (gfx1250) — matrix operand order repaired and checked by reading, not compiled, not run on hardware
//
#include <hip/hip_runtime.h>
#include <hip/hip_bf16.h>
#include <math.h>

// ---------------------------------------------------------------------------
// Problem constants (reference: B=4, S=2048, D=1024, E=8, R=128, H=4096)
// ---------------------------------------------------------------------------
constexpr int N_  = 4 * 2048;   // tokens
constexpr int D_  = 1024;
constexpr int E_  = 8;
constexpr int R_  = 128;
constexpr int H_  = 4096;
constexpr int NS_ = 2 * N_;     // (token, top-k slot) pairs
constexpr int KSTEP = 64;       // K elements staged to LDS per pipeline step

// ---------------------------------------------------------------------------
// CDNA5 WMMA types (wave32, v_wmma_f32_16x16x32_bf16)
// ---------------------------------------------------------------------------
typedef __bf16 bf16_t;
typedef __attribute__((ext_vector_type(8)))  __bf16 v8bf;
typedef __attribute__((ext_vector_type(16))) __bf16 v16bf;
typedef __attribute__((ext_vector_type(8)))  float  v8f;

union V16U { v16bf v; v8bf h[2]; };

// A fragment: 16x32 bf16 row-major. Lane owns row m = lane&15; hh = lane>>4.
// Lanes 0-15 hold K {0..15}, lanes 16-31 hold K {16..31}.
__device__ __forceinline__ v16bf load_a_row(const bf16_t* __restrict__ rowp,
                                            int k0, int hh) {
    const bf16_t* p = rowp + k0 + hh * 16;
    V16U r;
    r.h[0] = *(const v8bf*)(p);
    r.h[1] = *(const v8bf*)(p + 8);
    return r.v;
}

// B fragment from pre-transposed weight ([col][K] row-major) in GLOBAL memory.
__device__ __forceinline__ v16bf load_b_col(const bf16_t* __restrict__ Wt,
                                            int ldb, int col0, int k0, int lane) {
    const int hh = lane >> 4;
    const int n  = lane & 15;
    const bf16_t* p = Wt + (size_t)(col0 + n) * ldb + k0 + hh * 16;
    V16U r;
    r.h[0] = *(const v8bf*)(p);
    r.h[1] = *(const v8bf*)(p + 8);
    return r.v;
}

// B fragment from an LDS-staged tile laid out [64 cols][KSTEP k] row-major.
__device__ __forceinline__ v16bf load_b_lds(const bf16_t* base,
                                            int coloff, int k0, int lane) {
    const int hh = lane >> 4;
    const int n  = lane & 15;
    const bf16_t* p = base + (coloff + n) * KSTEP + k0 + hh * 16;
    V16U r;
    r.h[0] = *(const v8bf*)(p);
    r.h[1] = *(const v8bf*)(p + 8);
    return r.v;
}

__device__ __forceinline__ v8f wmma_bf16(v16bf a, v16bf b, v8f c) {
    return __builtin_amdgcn_wmma_f32_16x16x32_bf16(false, a, false, b,
                                                   (short)0, c, false, false);
}

// ---------------------------------------------------------------------------
// CDNA5 async global->LDS staging (ASYNCcnt-tracked)
// ---------------------------------------------------------------------------
#if __has_builtin(__builtin_amdgcn_s_wait_asynccnt)
#define WAIT_ASYNC(n) __builtin_amdgcn_s_wait_asynccnt(n)
#else
#define WAIT_ASYNC(n) asm volatile("s_wait_asynccnt %0" :: "i"(n) : "memory")
#endif

// 8 bytes per lane: LDS[lds_off] = MEM[gaddr]  (GLOBAL_LOAD_ASYNC_TO_LDS_B64)
__device__ __forceinline__ void async_g2l_b64(unsigned lds_off, const void* gaddr) {
    asm volatile("global_load_async_to_lds_b64 %0, %1, off"
                 :: "v"(lds_off), "v"(gaddr) : "memory");
}

__device__ __forceinline__ unsigned lds_addr32(const void* p) {
    // generic LDS pointer = {shared aperture, lds offset[31:0]}
    return (unsigned)(unsigned long long)p;
}

// Stage a [64 cols] x [KSTEP k] bf16 tile (8 KB) from Wt ([col][K] row-major)
// into sbuf. 256 threads x 4 chunks of 8 B. Every wave issues exactly 4 asyncs.
__device__ __forceinline__ void stage_b_tile(const bf16_t* __restrict__ Wt,
                                             int ldb, int colB, int kk,
                                             bf16_t* sbuf, int tid) {
    #pragma unroll
    for (int ch = 0; ch < 4; ++ch) {
        int chunk = tid + ch * 256;       // 0..1023
        int ccol  = chunk >> 4;           // 0..63  (16 x 8B chunks per column)
        int j     = chunk & 15;           // 8B = 4 halves
        const bf16_t* g = Wt + (size_t)(colB + ccol) * ldb + kk + j * 4;
        async_g2l_b64(lds_addr32(&sbuf[ccol * KSTEP + j * 4]), g);
    }
}

// ---------------------------------------------------------------------------
// Elementwise fp32 -> bf16 convert (tokens)
// ---------------------------------------------------------------------------
__global__ void cvt_f32_bf16_kernel(const float* __restrict__ in,
                                    bf16_t* __restrict__ out, long n) {
    long i = (long)blockIdx.x * blockDim.x + threadIdx.x;
    if (i < n) out[i] = (bf16_t)in[i];
}

// Convert + transpose weights: in fp32 [E][K][NC] -> out bf16 [E][NC][K]
__global__ void cvt_transpose_kernel(const float* __restrict__ in,
                                     bf16_t* __restrict__ out,
                                     int K, int NC, long total) {
    long i = (long)blockIdx.x * blockDim.x + threadIdx.x;
    if (i >= total) return;
    long n = i % NC;
    long k = (i / NC) % K;
    long e = i / ((long)NC * K);
    out[((long)e * NC + n) * K + k] = (bf16_t)in[i];
}

// ---------------------------------------------------------------------------
// Routing GEMM: acts[t][e][r] = x @ w_route[e]   (dense over all experts)
// grid (R/32, N/64, E), block 256 (8 waves -> 64x32 tile)
// ---------------------------------------------------------------------------
__global__ __launch_bounds__(256) void route_gemm_kernel(
        const bf16_t* __restrict__ xb, const bf16_t* __restrict__ wrt,
        bf16_t* __restrict__ acts) {
    const int lane = threadIdx.x & 31;
    const int wave = threadIdx.x >> 5;
    const int hh   = lane >> 4;
    const int e    = blockIdx.z;
    const int row0 = blockIdx.y * 64 + (wave >> 1) * 16;
    const int col0 = blockIdx.x * 32 + (wave & 1) * 16;

    const bf16_t* Arow = xb + (size_t)(row0 + (lane & 15)) * D_;
    const bf16_t* Bw   = wrt + (size_t)e * R_ * D_;

    v8f c = {0.f, 0.f, 0.f, 0.f, 0.f, 0.f, 0.f, 0.f};
    for (int k = 0; k < D_; k += 32) {
        c = wmma_bf16(load_a_row(Arow, k, hh),
                      load_b_col(Bw, D_, col0, k, lane), c);
    }
    const int n = col0 + (lane & 15);
    #pragma unroll
    for (int i = 0; i < 8; ++i) {
        int t = row0 + i + hh * 8;
        acts[((size_t)t * E_ + e) * R_ + n] = (bf16_t)c[i];
    }
}

// ---------------------------------------------------------------------------
// norms[t][e] = ||acts[t][e][:]||  (fp32, deterministic)
// ---------------------------------------------------------------------------
__global__ void norm_kernel(const bf16_t* __restrict__ acts,
                            float* __restrict__ norms_out) {
    int i = blockIdx.x * blockDim.x + threadIdx.x;   // t*E+e
    if (i >= N_ * E_) return;
    const bf16_t* p = acts + (size_t)i * R_;
    float s = 0.f;
    #pragma unroll 8
    for (int r = 0; r < R_; ++r) { float v = (float)p[r]; s += v * v; }
    norms_out[i] = sqrtf(s);
}

// ---------------------------------------------------------------------------
// Softmax + top-2 + combine weights + expert gather lists
// ---------------------------------------------------------------------------
__global__ void router_kernel(const float* __restrict__ norms,
                              float* __restrict__ probs,
                              float* __restrict__ combine,
                              int* __restrict__ sel,
                              int* __restrict__ counts,
                              int* __restrict__ lists) {
    int t = blockIdx.x * blockDim.x + threadIdx.x;
    if (t >= N_) return;
    float p[E_];
    float m = -1e30f;
    #pragma unroll
    for (int e = 0; e < E_; ++e) { p[e] = norms[t * E_ + e]; m = fmaxf(m, p[e]); }
    float sum = 0.f;
    #pragma unroll
    for (int e = 0; e < E_; ++e) { p[e] = __expf(p[e] - m); sum += p[e]; }
    float inv = 1.f / sum;
    #pragma unroll
    for (int e = 0; e < E_; ++e) {
        p[e] *= inv;
        probs[t * E_ + e]   = p[e];
        combine[t * E_ + e] = 0.f;
    }
    int i0 = 0;
    #pragma unroll
    for (int e = 1; e < E_; ++e) if (p[e] > p[i0]) i0 = e;   // ties -> lowest idx
    int i1 = (i0 == 0) ? 1 : 0;
    #pragma unroll
    for (int e = 0; e < E_; ++e) if (e != i0 && p[e] > p[i1]) i1 = e;
    float w0 = p[i0], w1 = p[i1], wn = w0 + w1;
    combine[t * E_ + i0] = w0 / wn;
    combine[t * E_ + i1] = w1 / wn;
    sel[t * 2 + 0] = i0;
    sel[t * 2 + 1] = i1;
    int pos = atomicAdd(&counts[i0], 1); lists[i0 * N_ + pos] = t * 2 + 0;
    pos     = atomicAdd(&counts[i1], 1); lists[i1 * N_ + pos] = t * 2 + 1;
}

// ---------------------------------------------------------------------------
// Expert up-projection on gathered tokens:
//   h[slot][col] = (x[t] @ w3[e])[col] * silu((acts[t][e] @ wr2h[e])[col])
// grid (H/64, N/128, E), block 256 = 8 waves (4x2), wave tile 32x32.
// w3 tile is double-buffer staged to LDS with async loads; gate GEMM (K=R)
// uses direct global loads.
// ---------------------------------------------------------------------------
__global__ __launch_bounds__(256) void ffn_up_kernel(
        const bf16_t* __restrict__ xb, const bf16_t* __restrict__ acts,
        const bf16_t* __restrict__ w3t, const bf16_t* __restrict__ wr2ht,
        const int* __restrict__ counts, const int* __restrict__ lists,
        bf16_t* __restrict__ hbuf) {
    __shared__ __align__(16) bf16_t sB[2][64 * KSTEP];   // 2 x 8 KB
    const int e   = blockIdx.z;
    const int cnt = counts[e];
    if ((int)blockIdx.y * 128 >= cnt) return;

    const int tid  = threadIdx.x;
    const int lane = tid & 31;
    const int wave = tid >> 5;
    const int hh   = lane >> 4;
    const int m    = lane & 15;
    const int row0 = blockIdx.y * 128 + (wave >> 1) * 32;
    const int colB = blockIdx.x * 64;          // block col base in H
    const int col0 = (wave & 1) * 32;          // wave col offset inside tile

    const int r0 = row0 + m, r1 = row0 + 16 + m;
    const int en0 = (r0 < cnt) ? lists[e * N_ + r0] : lists[e * N_];
    const int en1 = (r1 < cnt) ? lists[e * N_ + r1] : lists[e * N_];
    const int t0 = en0 >> 1, t1 = en1 >> 1;

    const bf16_t* Bu = w3t + (size_t)e * H_ * D_;         // [H][D]
    // prime the async pipeline with u-tile 0
    stage_b_tile(Bu, D_, colB, 0, sB[0], tid);

    // ---- gate GEMM: g = acts[t][e] @ wr2h[e], K = R (direct loads) ----
    const bf16_t* Ag0 = acts + ((size_t)t0 * E_ + e) * R_;
    const bf16_t* Ag1 = acts + ((size_t)t1 * E_ + e) * R_;
    const bf16_t* Bg  = wr2ht + (size_t)e * H_ * R_;      // [H][R]
    v8f g[2][2] = {};
    for (int k = 0; k < R_; k += 32) {
        v16bf a0 = load_a_row(Ag0, k, hh);
        v16bf a1 = load_a_row(Ag1, k, hh);
        v16bf b0 = load_b_col(Bg, R_, colB + col0,      k, lane);
        v16bf b1 = load_b_col(Bg, R_, colB + col0 + 16, k, lane);
        g[0][0] = wmma_bf16(a0, b0, g[0][0]);
        g[0][1] = wmma_bf16(a0, b1, g[0][1]);
        g[1][0] = wmma_bf16(a1, b0, g[1][0]);
        g[1][1] = wmma_bf16(a1, b1, g[1][1]);
    }

    // ---- up GEMM: u = x[t] @ w3[e], K = D, LDS double-buffered ----
    const bf16_t* Au0 = xb + (size_t)t0 * D_;
    const bf16_t* Au1 = xb + (size_t)t1 * D_;
    v8f u[2][2] = {};
    const int T = D_ / KSTEP;                 // 16 pipeline steps
    for (int t = 0; t < T; ++t) {
        if (t + 1 < T) {
            stage_b_tile(Bu, D_, colB, (t + 1) * KSTEP, sB[(t + 1) & 1], tid);
            WAIT_ASYNC(4);                    // tile t complete, t+1 in flight
        } else {
            WAIT_ASYNC(0);
        }
        __syncthreads();
        const bf16_t* bb = sB[t & 1];
        const int kk = t * KSTEP;
        #pragma unroll
        for (int k0 = 0; k0 < KSTEP; k0 += 32) {
            v16bf a0 = load_a_row(Au0, kk + k0, hh);
            v16bf a1 = load_a_row(Au1, kk + k0, hh);
            v16bf b0 = load_b_lds(bb, col0,      k0, lane);
            v16bf b1 = load_b_lds(bb, col0 + 16, k0, lane);
            u[0][0] = wmma_bf16(a0, b0, u[0][0]);
            u[0][1] = wmma_bf16(a0, b1, u[0][1]);
            u[1][0] = wmma_bf16(a1, b0, u[1][0]);
            u[1][1] = wmma_bf16(a1, b1, u[1][1]);
        }
        __syncthreads();                      // done reading sB[t&1]
    }

    // ---- h = u * silu(g), scatter to per-slot rows ----
    #pragma unroll
    for (int s2 = 0; s2 < 2; ++s2) {
        #pragma unroll
        for (int ct = 0; ct < 2; ++ct) {
            #pragma unroll
            for (int i = 0; i < 8; ++i) {
                int gi = row0 + s2 * 16 + i + hh * 8;
                if (gi < cnt) {
                    int en = lists[e * N_ + gi];
                    float gv = g[s2][ct][i];
                    float sg = 1.0f / (1.0f + __expf(-gv));
                    float hv = u[s2][ct][i] * (gv * sg);
                    hbuf[(size_t)en * H_ + colB + col0 + ct * 16 + m] = (bf16_t)hv;
                }
            }
        }
    }
}

// ---------------------------------------------------------------------------
// Expert down-projection:
//   out_slot[s][t][col] = (h[slot] @ wh2d[e])[col] * combine[t][e]
// grid (D/64, N/128, E), wave tile 32x32, weight tile async-staged to LDS.
// ---------------------------------------------------------------------------
__global__ __launch_bounds__(256) void ffn_down_kernel(
        const bf16_t* __restrict__ hbuf, const bf16_t* __restrict__ wh2dt,
        const int* __restrict__ counts, const int* __restrict__ lists,
        const float* __restrict__ combine, float* __restrict__ out_slot) {
    __shared__ __align__(16) bf16_t sB[2][64 * KSTEP];   // 2 x 8 KB
    const int e   = blockIdx.z;
    const int cnt = counts[e];
    if ((int)blockIdx.y * 128 >= cnt) return;

    const int tid  = threadIdx.x;
    const int lane = tid & 31;
    const int wave = tid >> 5;
    const int hh   = lane >> 4;
    const int m    = lane & 15;
    const int row0 = blockIdx.y * 128 + (wave >> 1) * 32;
    const int colB = blockIdx.x * 64;          // block col base in D
    const int col0 = (wave & 1) * 32;

    const int r0 = row0 + m, r1 = row0 + 16 + m;
    const int en0 = (r0 < cnt) ? lists[e * N_ + r0] : lists[e * N_];
    const int en1 = (r1 < cnt) ? lists[e * N_ + r1] : lists[e * N_];
    const bf16_t* A0 = hbuf + (size_t)en0 * H_;
    const bf16_t* A1 = hbuf + (size_t)en1 * H_;

    const bf16_t* Bd = wh2dt + (size_t)e * D_ * H_;       // [D][H]
    stage_b_tile(Bd, H_, colB, 0, sB[0], tid);

    v8f c[2][2] = {};
    const int T = H_ / KSTEP;                 // 64 pipeline steps
    for (int t = 0; t < T; ++t) {
        if (t + 1 < T) {
            stage_b_tile(Bd, H_, colB, (t + 1) * KSTEP, sB[(t + 1) & 1], tid);
            WAIT_ASYNC(4);
        } else {
            WAIT_ASYNC(0);
        }
        __syncthreads();
        const bf16_t* bb = sB[t & 1];
        const int kk = t * KSTEP;
        #pragma unroll
        for (int k0 = 0; k0 < KSTEP; k0 += 32) {
            v16bf a0 = load_a_row(A0, kk + k0, hh);
            v16bf a1 = load_a_row(A1, kk + k0, hh);
            v16bf b0 = load_b_lds(bb, col0,      k0, lane);
            v16bf b1 = load_b_lds(bb, col0 + 16, k0, lane);
            c[0][0] = wmma_bf16(a0, b0, c[0][0]);
            c[0][1] = wmma_bf16(a0, b1, c[0][1]);
            c[1][0] = wmma_bf16(a1, b0, c[1][0]);
            c[1][1] = wmma_bf16(a1, b1, c[1][1]);
        }
        __syncthreads();
    }

    #pragma unroll
    for (int s2 = 0; s2 < 2; ++s2) {
        #pragma unroll
        for (int ct = 0; ct < 2; ++ct) {
            #pragma unroll
            for (int i = 0; i < 8; ++i) {
                int gi = row0 + s2 * 16 + i + hh * 8;
                if (gi < cnt) {
                    int en = lists[e * N_ + gi];
                    int tt = en >> 1, sl = en & 1;
                    float w = combine[tt * E_ + e];
                    out_slot[((size_t)sl * N_ + tt) * D_ + colB + col0 + ct * 16 + m] =
                        c[s2][ct][i] * w;
                }
            }
        }
    }
}

// final = slot0 + slot1 (each slot written exactly once -> deterministic)
__global__ void combine_slots_kernel(const float* __restrict__ out_slot,
                                     float* __restrict__ out) {
    long i = (long)blockIdx.x * blockDim.x + threadIdx.x;
    if (i < (long)N_ * D_) out[i] = out_slot[i] + out_slot[(long)N_ * D_ + i];
}

// ---------------------------------------------------------------------------
// Load-balancing loss reductions (fixed-order tree reductions)
// ---------------------------------------------------------------------------
__global__ void rppe_kernel(const float* __restrict__ probs,
                            float* __restrict__ rppe) {
    int e = blockIdx.x;
    __shared__ float sm[256];
    float s = 0.f;
    for (int t = threadIdx.x; t < N_; t += 256) s += probs[t * E_ + e];
    sm[threadIdx.x] = s;
    __syncthreads();
    for (int w = 128; w > 0; w >>= 1) {
        if ((int)threadIdx.x < w) sm[threadIdx.x] += sm[threadIdx.x + w];
        __syncthreads();
    }
    if (threadIdx.x == 0) rppe[e] = sm[0] / (float)N_;
}

__global__ void tpe_kernel(const int* __restrict__ sel,
                           float* __restrict__ tpe) {
    int ke = blockIdx.x;          // k*E + e
    int k = ke / E_, e = ke % E_;
    __shared__ int sm[256];
    int c = 0;
    for (int t = threadIdx.x; t < N_; t += 256) c += (sel[t * 2 + k] == e);
    sm[threadIdx.x] = c;
    __syncthreads();
    for (int w = 128; w > 0; w >>= 1) {
        if ((int)threadIdx.x < w) sm[threadIdx.x] += sm[threadIdx.x + w];
        __syncthreads();
    }
    if (threadIdx.x == 0) tpe[ke] = (float)sm[0] / (float)N_;
}

__global__ void bl_final_kernel(const float* __restrict__ tpe,
                                const float* __restrict__ rppe,
                                float* __restrict__ out_bl) {
    if (threadIdx.x == 0 && blockIdx.x == 0) {
        float bl = 0.f;
        for (int k = 0; k < 2; ++k)
            for (int e = 0; e < E_; ++e) bl += tpe[k * E_ + e] * rppe[e];
        out_bl[0] = bl * (float)E_;
    }
}

// ---------------------------------------------------------------------------
// Host launcher
// ---------------------------------------------------------------------------
extern "C" void kernel_launch(void* const* d_in, const int* in_sizes, int n_in,
                              void* d_out, int out_size, void* d_ws, size_t ws_size,
                              hipStream_t stream) {
    (void)in_sizes; (void)n_in; (void)out_size; (void)ws_size;

    const float* x      = (const float*)d_in[0];   // [N, D]
    const float* w_rt   = (const float*)d_in[1];   // [E, D, R]
    const float* w3     = (const float*)d_in[2];   // [E, D, H]
    const float* wr2h   = (const float*)d_in[3];   // [E, R, H]
    const float* wh2d   = (const float*)d_in[4];   // [E, H, D]

    float* out       = (float*)d_out;              // final [N*D] | norms [N*E] | bl [1]
    float* norms_out = out + (size_t)N_ * D_;
    float* bl_out    = norms_out + (size_t)N_ * E_;

    // --- workspace carve-up (requires ~384 MB) ---
    char* ws = (char*)d_ws;
    size_t off = 0;
    auto carve = [&](size_t bytes) -> char* {
        char* p = ws + off;
        off = (off + bytes + 255) & ~(size_t)255;
        return p;
    };
    bf16_t* xb      = (bf16_t*)carve((size_t)N_ * D_ * 2);
    bf16_t* wrt     = (bf16_t*)carve((size_t)E_ * R_ * D_ * 2);   // [E][R][D]
    bf16_t* w3t     = (bf16_t*)carve((size_t)E_ * H_ * D_ * 2);   // [E][H][D]
    bf16_t* wr2ht   = (bf16_t*)carve((size_t)E_ * H_ * R_ * 2);   // [E][H][R]
    bf16_t* wh2dt   = (bf16_t*)carve((size_t)E_ * D_ * H_ * 2);   // [E][D][H]
    bf16_t* acts    = (bf16_t*)carve((size_t)N_ * E_ * R_ * 2);   // [N][E][R]
    bf16_t* hbuf    = (bf16_t*)carve((size_t)NS_ * H_ * 2);       // [2N][H]
    float*  outslot = (float*) carve((size_t)2 * N_ * D_ * 4);    // [2][N][D]
    float*  probs   = (float*) carve((size_t)N_ * E_ * 4);
    float*  combine = (float*) carve((size_t)N_ * E_ * 4);
    int*    sel     = (int*)   carve((size_t)NS_ * 4);
    int*    counts  = (int*)   carve((size_t)E_ * 4);
    int*    lists   = (int*)   carve((size_t)E_ * N_ * 4);
    float*  rppe    = (float*) carve((size_t)E_ * 4);
    float*  tpe     = (float*) carve((size_t)2 * E_ * 4);

    const int TB = 256;
    auto blocks = [&](long n) { return (unsigned)((n + TB - 1) / TB); };

    // 1) fp32 -> bf16 tokens; weights converted + transposed to [col][K]
    cvt_f32_bf16_kernel<<<blocks((long)N_ * D_), TB, 0, stream>>>(x, xb, (long)N_ * D_);
    cvt_transpose_kernel<<<blocks((long)E_ * D_ * R_), TB, 0, stream>>>(w_rt, wrt,  D_, R_, (long)E_ * D_ * R_);
    cvt_transpose_kernel<<<blocks((long)E_ * D_ * H_), TB, 0, stream>>>(w3,   w3t,  D_, H_, (long)E_ * D_ * H_);
    cvt_transpose_kernel<<<blocks((long)E_ * R_ * H_), TB, 0, stream>>>(wr2h, wr2ht, R_, H_, (long)E_ * R_ * H_);
    cvt_transpose_kernel<<<blocks((long)E_ * H_ * D_), TB, 0, stream>>>(wh2d, wh2dt, H_, D_, (long)E_ * H_ * D_);

    // 2) dense routing GEMM (WMMA) + fp32 norms
    route_gemm_kernel<<<dim3(R_ / 32, N_ / 64, E_), TB, 0, stream>>>(xb, wrt, acts);
    norm_kernel<<<blocks((long)N_ * E_), TB, 0, stream>>>(acts, norms_out);

    // 3) softmax / top-2 / combine / gather lists
    hipMemsetAsync(counts, 0, E_ * sizeof(int), stream);
    router_kernel<<<blocks(N_), TB, 0, stream>>>(norms_out, probs, combine, sel, counts, lists);

    // 4) sparse expert FFN on gathered tokens (WMMA + async LDS staging)
    ffn_up_kernel<<<dim3(H_ / 64, N_ / 128, E_), TB, 0, stream>>>(
        xb, acts, w3t, wr2ht, counts, lists, hbuf);
    ffn_down_kernel<<<dim3(D_ / 64, N_ / 128, E_), TB, 0, stream>>>(
        hbuf, wh2dt, counts, lists, combine, outslot);
    combine_slots_kernel<<<blocks((long)N_ * D_), TB, 0, stream>>>(outslot, out);

    // 5) load-balancing loss
    rppe_kernel<<<E_, TB, 0, stream>>>(probs, rppe);
    tpe_kernel<<<2 * E_, TB, 0, stream>>>(sel, tpe);
    bl_final_kernel<<<1, 32, 0, stream>>>(tpe, rppe, bl_out);
}
